// IOTransformer_90924457656391
// MI455X (gfx1250) — compile-verified, matrix operand order checked by reading, not run on hardware
//
#include <hip/hip_runtime.h>
#include <hip/hip_bf16.h>
#include <math.h>

typedef _Float16 v16h __attribute__((ext_vector_type(16)));
typedef _Float16 v8h  __attribute__((ext_vector_type(8)));
typedef float    v8f  __attribute__((ext_vector_type(8)));
typedef int      v4i_ __attribute__((ext_vector_type(4)));

#define D_  1024
#define T_  1024
#define B_  4
#define H_  16
#define L_  4
#define MROWS (B_*T_)      // 4096
#define NHEAD 96           // 64 act + 32 time

#define WMMA_F16(a,b,c) __builtin_amdgcn_wmma_f32_16x16x32_f16(false,(a),false,(b),(short)0,(c),false,false)

// ---- CDNA5 async load to LDS (ASYNCcnt-tracked, bypasses VGPRs) ----
__device__ __forceinline__ void async_copy_b128(void* lds, const void* gmem) {
#if __has_builtin(__builtin_amdgcn_global_load_async_to_lds_b128)
  typedef __attribute__((address_space(1))) v4i_* gptr_t;
  typedef __attribute__((address_space(3))) v4i_* lptr_t;
  __builtin_amdgcn_global_load_async_to_lds_b128((gptr_t)(void*)gmem, (lptr_t)lds, 0, 0);
#else
  uint32_t l = (uint32_t)(size_t)(__attribute__((address_space(3))) void*)lds;
  uint64_t g = (uint64_t)(size_t)gmem;
  asm volatile("global_load_async_to_lds_b128 %0, %1, off" :: "v"(l), "v"(g) : "memory");
#endif
}

__device__ __forceinline__ void wait_async0() {
#if __has_builtin(__builtin_amdgcn_s_wait_asynccnt)
  __builtin_amdgcn_s_wait_asynccnt(0);
#else
  asm volatile("s_wait_asynccnt 0x0" ::: "memory");
#endif
}

// ---- CDNA5 WMMA fragment loaders (16-bit A layout, ISA 7.12.2) ----
// lane L: row = L&15, khalf = L>>4 ; VGPR0-3 hold K=khalf*8..+7, VGPR4-7 hold K=16+khalf*8..+7
__device__ __forceinline__ v16h frag_a_h(const _Float16* base, int ld, int lane) {
  int row = lane & 15;
  int kh  = (lane >> 4) << 3;
  const _Float16* p = base + row * ld + kh;
  v8h lo = *(const v8h*)p;
  v8h hi = *(const v8h*)(p + 16);
  v16h f;
#pragma unroll
  for (int j = 0; j < 8; ++j) { f[j] = lo[j]; f[8 + j] = hi[j]; }
  return f;
}

__device__ __forceinline__ v16h frag_a_f32(const float* base, int ld, int lane) {
  int row = lane & 15;
  int kh  = (lane >> 4) << 3;
  const float* p = base + row * ld + kh;
  v16h f;
#pragma unroll
  for (int j = 0; j < 8; ++j) { f[j] = (_Float16)p[j]; f[8 + j] = (_Float16)p[16 + j]; }
  return f;
}

// =============== embedding + posenc + feature projections ===============
__global__ __launch_bounds__(256) void embed_kernel(
    const int* __restrict__ tokens, const int* __restrict__ cat_feats,
    const float* __restrict__ num_feats, const float* __restrict__ time_feats,
    const int* __restrict__ task_ids, const float* __restrict__ E,
    const float* __restrict__ ce0, const float* __restrict__ ce1,
    const float* __restrict__ Wn, const float* __restrict__ bn,
    const float* __restrict__ Wt, const float* __restrict__ bt,
    const float* __restrict__ task_embed, float* __restrict__ x) {
  int row = blockIdx.x;               // b*T + t
  int b = row >> 10, t = row & 1023;
  int tok = tokens[row];
  int c0 = cat_feats[row * 2 + 0], c1 = cat_feats[row * 2 + 1];
  int task = task_ids[b];
  float nf[4], tf[6];
#pragma unroll
  for (int j = 0; j < 4; ++j) nf[j] = num_feats[row * 4 + j];
#pragma unroll
  for (int j = 0; j < 6; ++j) tf[j] = time_feats[row * 6 + j];
  for (int d = threadIdx.x; d < D_; d += 256) {
    float v = E[(size_t)tok * D_ + d] + ce0[(size_t)c0 * D_ + d] + ce1[(size_t)c1 * D_ + d]
            + bn[d] + bt[d] + task_embed[(size_t)task * D_ + d];
#pragma unroll
    for (int j = 0; j < 4; ++j) v += nf[j] * Wn[j * D_ + d];
#pragma unroll
    for (int j = 0; j < 6; ++j) v += tf[j] * Wt[j * D_ + d];
    int i = d >> 1;
    float ang = (float)t * __expf(-((float)(2 * i) / (float)D_) * 9.210340371976184f); // ln(1e4)
    v += (d & 1) ? __cosf(ang) : __sinf(ang);
    x[(size_t)row * D_ + d] = v;
  }
}

// =============== LayerNorm (f32 in -> f16 out) ===============
__global__ __launch_bounds__(256) void ln_kernel(const float* __restrict__ x,
    const float* __restrict__ g, const float* __restrict__ bb, _Float16* __restrict__ y) {
  __shared__ float red[256];
  int row = blockIdx.x, t = threadIdx.x;
  const float* xr = x + (size_t)row * D_;
  float v[4];
  float s = 0.f;
#pragma unroll
  for (int i = 0; i < 4; ++i) { v[i] = xr[t + 256 * i]; s += v[i]; }
  red[t] = s; __syncthreads();
  for (int off = 128; off > 0; off >>= 1) { if (t < off) red[t] += red[t + off]; __syncthreads(); }
  float mean = red[0] * (1.f / 1024.f);
  __syncthreads();
  s = 0.f;
#pragma unroll
  for (int i = 0; i < 4; ++i) { float d = v[i] - mean; s += d * d; }
  red[t] = s; __syncthreads();
  for (int off = 128; off > 0; off >>= 1) { if (t < off) red[t] += red[t + off]; __syncthreads(); }
  float rs = rsqrtf(red[0] * (1.f / 1024.f) + 1e-5f);
  _Float16* yr = y + (size_t)row * D_;
#pragma unroll
  for (int i = 0; i < 4; ++i) { int d = t + 256 * i; yr[d] = (_Float16)((v[i] - mean) * rs * g[d] + bb[d]); }
}

// =============== weight transpose + f32->f16 (W[K,N] -> Wt[N,K]) ===============
__global__ __launch_bounds__(256) void transpose_cvt(const float* __restrict__ W,
    _Float16* __restrict__ Wt, int K, int N) {
  size_t idx = (size_t)blockIdx.x * 256 + threadIdx.x;
  if (idx >= (size_t)K * N) return;
  int k = (int)(idx / N), n = (int)(idx % N);
  Wt[(size_t)n * K + k] = (_Float16)W[idx];
}

// =============== WMMA GEMM: C[M,N] = act(A[M,K] @ Bt[N,K]^T + bias) (+resid) ===============
// 256 thr / 8 waves. 128x128 block tile, each wave owns 32x64 (8 accumulators).
// Double-buffered LDS fed by async-to-LDS DMA; 64 WMMAs per block per K-step.
__global__ __launch_bounds__(256) void gemm_wmma(
    const _Float16* __restrict__ A, const _Float16* __restrict__ Bt,
    const float* __restrict__ bias, const float* __restrict__ resid,
    float* __restrict__ Cf, _Float16* __restrict__ Ch,
    int M, int N, int K, int relu) {
  __shared__ _Float16 As[2][128 * 40];
  __shared__ _Float16 Bs[2][128 * 40];
  int m0 = blockIdx.y * 128, n0 = blockIdx.x * 128;
  int t = threadIdx.x, w = t >> 5, lane = t & 31;
  int wm = w & 3, wn = w >> 2;              // wave tile: rows 32*wm, cols 64*wn
  // loader geometry: 512 b128 chunks per operand stage; 2 per thread
  int lr0 = t >> 2, lc0 = (t & 3) * 8;      // chunk 0: rows 0..63
  int lr1 = lr0 + 64, lc1 = lc0;            // chunk 1: rows 64..127

  v8f acc[2][4] = {};

  // stage 0 prefetch
  {
    async_copy_b128(&As[0][lr0 * 40 + lc0], &A[(size_t)(m0 + lr0) * K + lc0]);
    async_copy_b128(&As[0][lr1 * 40 + lc1], &A[(size_t)(m0 + lr1) * K + lc1]);
    async_copy_b128(&Bs[0][lr0 * 40 + lc0], &Bt[(size_t)(n0 + lr0) * K + lc0]);
    async_copy_b128(&Bs[0][lr1 * 40 + lc1], &Bt[(size_t)(n0 + lr1) * K + lc1]);
  }
  wait_async0();
  __syncthreads();

  int nsteps = K >> 5;
  for (int s = 0; s < nsteps; ++s) {
    int cur = s & 1;
    if (s + 1 < nsteps) {                  // prefetch next stage while computing
      int k0 = (s + 1) << 5;
      async_copy_b128(&As[cur ^ 1][lr0 * 40 + lc0], &A[(size_t)(m0 + lr0) * K + k0 + lc0]);
      async_copy_b128(&As[cur ^ 1][lr1 * 40 + lc1], &A[(size_t)(m0 + lr1) * K + k0 + lc1]);
      async_copy_b128(&Bs[cur ^ 1][lr0 * 40 + lc0], &Bt[(size_t)(n0 + lr0) * K + k0 + lc0]);
      async_copy_b128(&Bs[cur ^ 1][lr1 * 40 + lc1], &Bt[(size_t)(n0 + lr1) * K + k0 + lc1]);
    }
    v16h a0 = frag_a_h(&As[cur][(wm * 32) * 40], 40, lane);
    v16h a1 = frag_a_h(&As[cur][(wm * 32 + 16) * 40], 40, lane);
    v16h bf[4];
#pragma unroll
    for (int j = 0; j < 4; ++j) bf[j] = frag_a_h(&Bs[cur][(wn * 64 + 16 * j) * 40], 40, lane);
#pragma unroll
    for (int j = 0; j < 4; ++j) {
      acc[0][j] = WMMA_F16(a0, bf[j], acc[0][j]);
      acc[1][j] = WMMA_F16(a1, bf[j], acc[1][j]);
    }
    wait_async0();
    __syncthreads();
  }

  int nin = lane & 15, mb = (lane >> 4) * 8;
#pragma unroll
  for (int ii = 0; ii < 2; ++ii) {
#pragma unroll
    for (int j = 0; j < 4; ++j) {
      int n = n0 + wn * 64 + 16 * j + nin;
      float bv = bias ? bias[n] : 0.f;
#pragma unroll
      for (int i = 0; i < 8; ++i) {
        int m = m0 + wm * 32 + 16 * ii + mb + i;
        float v = acc[ii][j][i] + bv;
        if (relu) v = v > 0.f ? v : 0.f;
        if (resid) v += resid[(size_t)m * N + n];
        if (Cf) Cf[(size_t)m * N + n] = v;
        if (Ch) Ch[(size_t)m * N + n] = (_Float16)v;
      }
    }
  }
}

// =============== fused causal attention, one (b,h,32-query) block ===============
// qkv: f16 [B*T, 3072] (q|k|v). scores kept in LDS (32x1024 f32 = 128KB of 320KB WGP LDS).
#define ATTN_SMEM (32*1024*4 + 32*72*2 + 32*72*2 + 64*40*2)
__global__ __launch_bounds__(128) void attn_kernel(const _Float16* __restrict__ qkv,
    const int* __restrict__ amask, _Float16* __restrict__ out) {
  extern __shared__ char smem[];
  float*    Sf  = (float*)smem;                          // [32][1024]
  _Float16* Qs  = (_Float16*)(smem + 32 * 1024 * 4);     // [32][72]
  _Float16* Ks  = Qs + 32 * 72;                          // [32][72]
  _Float16* Vts = Ks + 32 * 72;                          // [64][40] (V^T)
  int q0 = blockIdx.x * 32, h = blockIdx.y, b = blockIdx.z;
  int t = threadIdx.x, w = t >> 5, lane = t & 31;
  int tr = w & 1, tc = w >> 1;
  size_t base = (size_t)(b * T_) * 3072 + h * 64;
  { // load Q tile 32x64 (async DMA to LDS)
    int r = t >> 2, c = (t & 3) * 16;
    async_copy_b128(&Qs[r * 72 + c], &qkv[base + (size_t)(q0 + r) * 3072 + c]);
    async_copy_b128(&Qs[r * 72 + c + 8], &qkv[base + (size_t)(q0 + r) * 3072 + c + 8]);
  }
  wait_async0();
  __syncthreads();
  int nkb = (q0 >> 5) + 1;
  // ---- phase 1: scores ----
  for (int kb = 0; kb < nkb; ++kb) {
    { int r = t >> 2, c = (t & 3) * 16;
      async_copy_b128(&Ks[r * 72 + c], &qkv[base + 1024 + (size_t)(kb * 32 + r) * 3072 + c]);
      async_copy_b128(&Ks[r * 72 + c + 8], &qkv[base + 1024 + (size_t)(kb * 32 + r) * 3072 + c + 8]);
    }
    wait_async0();
    __syncthreads();
    v8f acc = {};
#pragma unroll
    for (int ks = 0; ks < 64; ks += 32) {        // S = Q @ K^T  (Bt == K)
      v16h aq = frag_a_h(&Qs[tr * 16 * 72 + ks], 72, lane);
      v16h bk = frag_a_h(&Ks[tc * 16 * 72 + ks], 72, lane);
      acc = WMMA_F16(aq, bk, acc);
    }
    int nin = lane & 15, mb = (lane >> 4) * 8;
    int gk = kb * 32 + tc * 16 + nin;
    float madd = (amask[b * T_ + gk] == 0) ? -1e9f : 0.f;
#pragma unroll
    for (int i = 0; i < 8; ++i) {
      int m = tr * 16 + mb + i;
      int gq = q0 + m;
      float v = acc[i] * 0.125f + madd + ((gk > gq) ? -1e9f : 0.f);
      Sf[m * 1024 + kb * 32 + tc * 16 + nin] = v;
    }
    __syncthreads();
  }
  // ---- phase 2: softmax over valid keys ----
  int nk = nkb * 32;
  {
    int row = t >> 2, q = t & 3;
    float mx = -3.0e38f;
    for (int c = q; c < nk; c += 4) mx = fmaxf(mx, Sf[row * 1024 + c]);
    mx = fmaxf(mx, __shfl_xor(mx, 1));
    mx = fmaxf(mx, __shfl_xor(mx, 2));
    float s = 0.f;
    for (int c = q; c < nk; c += 4) { float e = __expf(Sf[row * 1024 + c] - mx); Sf[row * 1024 + c] = e; s += e; }
    s += __shfl_xor(s, 1);
    s += __shfl_xor(s, 2);
    float inv = 1.f / s;
    for (int c = q; c < nk; c += 4) Sf[row * 1024 + c] *= inv;
  }
  __syncthreads();
  // ---- phase 3: O = P @ V ----
  v8f o0 = {}, o1 = {};
  for (int kb = 0; kb < nkb; ++kb) {
    { // load V transposed: Vts[d][j] = V[kb*32+j][d]
      int j = t >> 2, d0 = (t & 3) * 16;
      const _Float16* vp = &qkv[base + 2048 + (size_t)(kb * 32 + j) * 3072];
#pragma unroll
      for (int dd = 0; dd < 16; ++dd) Vts[(d0 + dd) * 40 + j] = vp[d0 + dd];
    }
    __syncthreads();
    v16h ap  = frag_a_f32(&Sf[tr * 16 * 1024 + kb * 32], 1024, lane);
    v16h bv0 = frag_a_h(&Vts[(tc * 32) * 40], 40, lane);
    v16h bv1 = frag_a_h(&Vts[(tc * 32 + 16) * 40], 40, lane);
    o0 = WMMA_F16(ap, bv0, o0);
    o1 = WMMA_F16(ap, bv1, o1);
    __syncthreads();
  }
  int nin = lane & 15, mb = (lane >> 4) * 8;
#pragma unroll
  for (int i = 0; i < 8; ++i) {
    int m = tr * 16 + mb + i;
    size_t orow = (size_t)(b * T_ + q0 + m) * 1024 + h * 64 + tc * 32;
    out[orow + nin]      = (_Float16)o0[i];
    out[orow + 16 + nin] = (_Float16)o1[i];
  }
}

// =============== head helpers ===============
__global__ __launch_bounds__(256) void build_wcomb(const float* __restrict__ W_next,
    const float* __restrict__ W_timeh, const float* __restrict__ E,
    const float* __restrict__ tsa, const float* __restrict__ tst, _Float16* __restrict__ Wc) {
  float spa = log1pf(__expf(tsa[0]));
  float spt = log1pf(__expf(tst[0]));
  int idx = blockIdx.x * 256 + threadIdx.x;      // 96*1024
  int c = idx >> 10, k = idx & 1023;
  float v;
  if (c < 64) v = W_next[k * 64 + c] + spa * E[(size_t)(4 + c) * D_ + k];
  else { int c2 = c - 64; v = W_timeh[k * 32 + c2] + spt * E[(size_t)(68 + c2) * D_ + k]; }
  Wc[idx] = (_Float16)v;
}

__global__ __launch_bounds__(256) void zero_f32(float* __restrict__ p, int n) {
  int i = blockIdx.x * 256 + threadIdx.x;
  if (i < n) p[i] = 0.f;
}

// M[b,c,:] = sum of h rows whose prev token == LABEL and token maps to class c
__global__ __launch_bounds__(256) void accum_m(const int* __restrict__ tokens,
    const _Float16* __restrict__ h, float* __restrict__ M) {
  int row = blockIdx.x;
  int b = row >> 10, k = row & 1023;
  if (k == 0) return;
  if (tokens[row - 1] != 3) return;              // LABEL_ID
  int tok = tokens[row];
  int c;
  if (tok >= 4 && tok < 68) c = tok - 4;
  else if (tok >= 68 && tok < 100) c = 64 + (tok - 68);
  else return;
  float* dst = M + (size_t)(b * NHEAD + c) * D_;
  const _Float16* src = h + (size_t)row * D_;
  for (int d = threadIdx.x; d < D_; d += 256) atomicAdd(&dst[d], (float)src[d]);
}

__global__ __launch_bounds__(256) void scale_m(const float* __restrict__ M,
    const float* __restrict__ csa, const float* __restrict__ cst, _Float16* __restrict__ Mh) {
  int idx = blockIdx.x * 256 + threadIdx.x;      // B*96*1024
  int c = (idx >> 10) % NHEAD;
  float sp = (c < 64) ? log1pf(__expf(csa[0])) : log1pf(__expf(cst[0]));
  Mh[idx] = (_Float16)(M[idx] * sp * 0.03125f);  // 1/sqrt(D)=1/32 folded in
}

// =============== fused head GEMM: both logit tensors at once ===============
// logits = h @ Wcomb + bias + loss_mask * (h @ M[b])   (M pre-scaled)
__global__ __launch_bounds__(256) void head_kernel(const _Float16* __restrict__ Hh,
    const _Float16* __restrict__ Wc, const _Float16* __restrict__ Mh,
    const float* __restrict__ b_next, const float* __restrict__ b_timeh,
    const int* __restrict__ loss_mask, float* __restrict__ out_act, float* __restrict__ out_time) {
  __shared__ _Float16 Hs[64 * 40];
  __shared__ _Float16 Ws[NHEAD * 40];
  __shared__ _Float16 Ms[NHEAD * 40];
  int q0 = blockIdx.x * 64, b = blockIdx.y;
  int t = threadIdx.x, w = t >> 5, lane = t & 31;
  int wr = w & 3, wc = w >> 2;
  const _Float16* Mb = Mh + (size_t)b * NHEAD * D_;
  v8f accW[3] = {}, accM[3] = {};
  for (int k0 = 0; k0 < D_; k0 += 32) {
    { int r = t >> 2, c = (t & 3) * 8;
      async_copy_b128(&Hs[r * 40 + c], &Hh[(size_t)(b * T_ + q0 + r) * D_ + k0 + c]); }
    for (int e = t; e < NHEAD * 4; e += 256) {
      int r = e >> 2, c = (e & 3) * 8;
      async_copy_b128(&Ws[r * 40 + c], &Wc[(size_t)r * D_ + k0 + c]);
      async_copy_b128(&Ms[r * 40 + c], &Mb[(size_t)r * D_ + k0 + c]);
    }
    wait_async0();
    __syncthreads();
    v16h a = frag_a_h(&Hs[wr * 16 * 40], 40, lane);
#pragma unroll
    for (int j = 0; j < 3; ++j) {
      v16h bw = frag_a_h(&Ws[(wc * 48 + j * 16) * 40], 40, lane);
      v16h bm = frag_a_h(&Ms[(wc * 48 + j * 16) * 40], 40, lane);
      accW[j] = WMMA_F16(a, bw, accW[j]);
      accM[j] = WMMA_F16(a, bm, accM[j]);
    }
    __syncthreads();
  }
  int nin = lane & 15, mb = (lane >> 4) * 8;
#pragma unroll
  for (int j = 0; j < 3; ++j) {
    int n = wc * 48 + j * 16 + nin;
#pragma unroll
    for (int i = 0; i < 8; ++i) {
      int q = q0 + wr * 16 + mb + i;
      float qm = loss_mask[b * T_ + q] ? 1.f : 0.f;
      float v = accW[j][i] + qm * accM[j][i];
      if (n < 64) out_act[(size_t)(b * T_ + q) * 64 + n] = v + b_next[n];
      else        out_time[(size_t)(b * T_ + q) * 32 + (n - 64)] = v + b_timeh[n - 64];
    }
  }
}

// =============================================================================
extern "C" void kernel_launch(void* const* d_in, const int* in_sizes, int n_in,
                              void* d_out, int out_size, void* d_ws, size_t ws_size,
                              hipStream_t stream) {
  const int*   tokens   = (const int*)d_in[0];
  const int*   cat      = (const int*)d_in[1];
  const float* numf     = (const float*)d_in[2];
  const float* timef    = (const float*)d_in[3];
  const int*   amask    = (const int*)d_in[4];
  const int*   lmask    = (const int*)d_in[5];
  const int*   task     = (const int*)d_in[6];
  const float* E        = (const float*)d_in[7];
  const float* ce0      = (const float*)d_in[8];
  const float* ce1      = (const float*)d_in[9];
  const float* Wn       = (const float*)d_in[10];
  const float* bn       = (const float*)d_in[11];
  const float* Wt       = (const float*)d_in[12];
  const float* bt       = (const float*)d_in[13];
  const float* Wqkv     = (const float*)d_in[14];
  const float* bqkv     = (const float*)d_in[15];
  const float* Wo       = (const float*)d_in[16];
  const float* bo       = (const float*)d_in[17];
  const float* ln1_g    = (const float*)d_in[18];
  const float* ln1_b    = (const float*)d_in[19];
  const float* W1       = (const float*)d_in[20];
  const float* b1       = (const float*)d_in[21];
  const float* W2       = (const float*)d_in[22];
  const float* b2       = (const float*)d_in[23];
  const float* ln2_g    = (const float*)d_in[24];
  const float* ln2_b    = (const float*)d_in[25];
  const float* lnf_g    = (const float*)d_in[26];
  const float* lnf_b    = (const float*)d_in[27];
  const float* W_next   = (const float*)d_in[28];
  const float* b_next   = (const float*)d_in[29];
  const float* W_timeh  = (const float*)d_in[30];
  const float* b_timeh  = (const float*)d_in[31];
  const float* task_emb = (const float*)d_in[32];
  const float* tsa      = (const float*)d_in[33];
  const float* tst      = (const float*)d_in[34];
  const float* csa      = (const float*)d_in[35];
  const float* cst      = (const float*)d_in[36];

  // ---- workspace layout ----
  char* ws = (char*)d_ws;
  size_t o = 0;
  auto alloc = [&](size_t bytes) -> char* {
    char* p = ws + o;
    o = (o + bytes + 255) & ~(size_t)255;
    return p;
  };
  float*    x      = (float*)   alloc((size_t)MROWS * D_ * 4);
  _Float16* hln    = (_Float16*)alloc((size_t)MROWS * D_ * 2);
  _Float16* qkv    = (_Float16*)alloc((size_t)MROWS * 3072 * 2);
  _Float16* attn   = (_Float16*)alloc((size_t)MROWS * D_ * 2);
  _Float16* ff     = (_Float16*)alloc((size_t)MROWS * 4096 * 2);
  _Float16* hh     = (_Float16*)alloc((size_t)MROWS * D_ * 2);
  _Float16* wqkv_t = (_Float16*)alloc((size_t)3072 * D_ * 2);
  _Float16* wo_t   = (_Float16*)alloc((size_t)D_ * D_ * 2);
  _Float16* w1_t   = (_Float16*)alloc((size_t)4096 * D_ * 2);
  _Float16* w2_t   = (_Float16*)alloc((size_t)D_ * 4096 * 2);
  _Float16* wcomb  = (_Float16*)alloc((size_t)NHEAD * D_ * 2);
  float*    Mf     = (float*)   alloc((size_t)B_ * NHEAD * D_ * 4);
  _Float16* Mh     = (_Float16*)alloc((size_t)B_ * NHEAD * D_ * 2);

  float* out_act  = (float*)d_out;
  float* out_time = out_act + (size_t)B_ * T_ * 64;

  (void)hipFuncSetAttribute((const void*)attn_kernel,
                            hipFuncAttributeMaxDynamicSharedMemorySize, ATTN_SMEM);

  // ---- embedding ----
  embed_kernel<<<MROWS, 256, 0, stream>>>(tokens, cat, numf, timef, task, E, ce0, ce1,
                                          Wn, bn, Wt, bt, task_emb, x);

  // ---- transformer layers ----
  for (int l = 0; l < L_; ++l) {
    transpose_cvt<<<(D_ * 3072) / 256, 256, 0, stream>>>(Wqkv + (size_t)l * D_ * 3072, wqkv_t, D_, 3072);
    transpose_cvt<<<(D_ * D_) / 256, 256, 0, stream>>>(Wo + (size_t)l * D_ * D_, wo_t, D_, D_);
    transpose_cvt<<<(D_ * 4096) / 256, 256, 0, stream>>>(W1 + (size_t)l * D_ * 4096, w1_t, D_, 4096);
    transpose_cvt<<<(4096 * D_) / 256, 256, 0, stream>>>(W2 + (size_t)l * 4096 * D_, w2_t, 4096, D_);

    ln_kernel<<<MROWS, 256, 0, stream>>>(x, ln1_g + l * D_, ln1_b + l * D_, hln);
    gemm_wmma<<<dim3(3072 / 128, MROWS / 128), 256, 0, stream>>>(
        hln, wqkv_t, bqkv + l * 3072, nullptr, nullptr, qkv, MROWS, 3072, D_, 0);

    attn_kernel<<<dim3(T_ / 32, H_, B_), 128, ATTN_SMEM, stream>>>(qkv, amask, attn);

    gemm_wmma<<<dim3(D_ / 128, MROWS / 128), 256, 0, stream>>>(
        attn, wo_t, bo + l * D_, x, x, nullptr, MROWS, D_, D_, 0);

    ln_kernel<<<MROWS, 256, 0, stream>>>(x, ln2_g + l * D_, ln2_b + l * D_, hln);
    gemm_wmma<<<dim3(4096 / 128, MROWS / 128), 256, 0, stream>>>(
        hln, w1_t, b1 + (size_t)l * 4096, nullptr, nullptr, ff, MROWS, 4096, D_, 1);
    gemm_wmma<<<dim3(D_ / 128, MROWS / 128), 256, 0, stream>>>(
        ff, w2_t, b2 + l * D_, x, x, nullptr, MROWS, D_, 4096, 0);
  }

  // ---- final norm + heads ----
  ln_kernel<<<MROWS, 256, 0, stream>>>(x, lnf_g, lnf_b, hh);
  build_wcomb<<<(NHEAD * D_) / 256, 256, 0, stream>>>(W_next, W_timeh, E, tsa, tst, wcomb);
  zero_f32<<<(B_ * NHEAD * D_) / 256, 256, 0, stream>>>(Mf, B_ * NHEAD * D_);
  accum_m<<<MROWS, 256, 0, stream>>>(tokens, hh, Mf);
  scale_m<<<(B_ * NHEAD * D_) / 256, 256, 0, stream>>>(Mf, csa, cst, Mh);
  head_kernel<<<dim3(T_ / 64, B_), 256, 0, stream>>>(hh, wcomb, Mh, b_next, b_timeh,
                                                     lmask, out_act, out_time);
}
